// SingleHeadAttention_75127567941725
// MI455X (gfx1250) — matrix operand order repaired and checked by reading, not compiled, not run on hardware
//
#include <hip/hip_runtime.h>

// ---------------------------------------------------------------------------
// CDNA5 (gfx1250, wave32) flash attention, v3:
//   K0: fp32 -> bf16 convert pre-pass (embedded + Wk/Wq/Wv), memory-bound.
//   K1: QKV projection: pure bf16 b128 loads -> v_wmma_f32_16x16x32_bf16.
//       Emits Q (pre-scaled 1/sqrt(A)), K transposed [B,A,T], V [B,T,A].
//   K2: causal flash attention with triangle-pair load balancing; Q.K^T and
//       P.V on the WMMA pipe, online softmax with wave32 shfl-xor, P relayout
//       through LDS (s_wait_dscnt), v_rcp normalization.
// ---------------------------------------------------------------------------

typedef __attribute__((ext_vector_type(16))) __bf16 v16bf;
typedef __attribute__((ext_vector_type(8)))  __bf16 v8bf;
typedef __attribute__((ext_vector_type(4)))  __bf16 v4bf;
typedef __attribute__((ext_vector_type(8)))  float  v8f;
typedef __attribute__((ext_vector_type(4)))  float  v4f;

namespace {
constexpr int   kB     = 4;
constexpr int   kT     = 4096;
constexpr int   kE     = 256;
constexpr int   kA     = 64;
constexpr int   kTok   = kB * kT;            // 16384 tokens
constexpr int   kQT    = kT / 16;            // 256 query tiles per batch
constexpr float kScale = 0.125f;             // 1/sqrt(A)
}

// round-to-nearest-even fp32 -> bf16
__device__ __forceinline__ __bf16 f2bf(float f) {
  unsigned u = __builtin_bit_cast(unsigned, f);
  unsigned r = u + 0x7FFFu + ((u >> 16) & 1u);
  unsigned short h = (unsigned short)(r >> 16);
  return __builtin_bit_cast(__bf16, h);
}

__device__ __forceinline__ v8f wmma_bf16(v16bf a, v16bf b, v8f c) {
  // (neg_a, A, neg_b, B, c_mod, C, reuse_a, reuse_b)
  return __builtin_amdgcn_wmma_f32_16x16x32_bf16(false, a, false, b,
                                                 (short)0, c, false, false);
}

__device__ __forceinline__ v16bf cat8(v8bf lo, v8bf hj) {
  return __builtin_shufflevector(lo, hj, 0, 1, 2, 3, 4, 5, 6, 7,
                                         8, 9, 10, 11, 12, 13, 14, 15);
}

// ---------------------------------------------------------------------------
// Kernel 0: bulk fp32 -> bf16 (4 elements per thread, b128 in / b64 out).
// ---------------------------------------------------------------------------
__global__ __launch_bounds__(256) void to_bf16_kernel(
    const float* __restrict__ src, __bf16* __restrict__ dst, int n)
{
  const int i = (blockIdx.x * 256 + threadIdx.x) * 4;
  if (i >= n) return;
  v4f f = *(const v4f*)(src + i);
  v4bf o;
  #pragma unroll
  for (int e = 0; e < 4; ++e) o[e] = f2bf(f[e]);
  *(v4bf*)(dst + i) = o;
}

// ---------------------------------------------------------------------------
// Kernel 1: q = (emb@Wq + bq)*scale ; k = emb@Wk + bk (stored [B,A,T]) ;
//           v = emb@Wv + bv.  One 16-token M-tile per wave, K-loop over E.
//           All operands pre-converted bf16: inner loop is loads + WMMA only.
// ---------------------------------------------------------------------------
__global__ __launch_bounds__(128) void qkv_proj_kernel(
    const __bf16* __restrict__ embB,
    const __bf16* __restrict__ wkB, const __bf16* __restrict__ wqB,
    const __bf16* __restrict__ wvB,
    const float* __restrict__ bk, const float* __restrict__ bq,
    const float* __restrict__ bv,
    __bf16* __restrict__ qOut, __bf16* __restrict__ kTOut,
    __bf16* __restrict__ vOut)
{
  const int lane = threadIdx.x & 31;
  const int wave = threadIdx.x >> 5;
  const int hi   = lane >> 4;                 // half-wave select
  const int nrow = lane & 15;
  const long mbase = ((long)blockIdx.x * 4 + wave) * 16;   // token base
  const int  b     = (int)(mbase / kT);
  const int  tbase = (int)(mbase % kT);

  v8f accK[4], accQ[4], accV[4];
  #pragma unroll
  for (int nt = 0; nt < 4; ++nt)
    #pragma unroll
    for (int e = 0; e < 8; ++e) { accK[nt][e] = 0.f; accQ[nt][e] = 0.f; accV[nt][e] = 0.f; }

  // A-fragment K offsets for 16-bit 16x32 A (ISA 7.12.2)
  const int klo = hi * 8;        // halves 0..7
  const int khi = 16 + hi * 8;   // halves 8..15
  const __bf16* arow = embB + (mbase + nrow) * kE;

  for (int c = 0; c < kE / 32; ++c) {
    const v16bf af = cat8(*(const v8bf*)(arow + c * 32 + klo),
                          *(const v8bf*)(arow + c * 32 + khi));
    #pragma unroll
    for (int nt = 0; nt < 4; ++nt) {
      v16bf bK, bQ, bV;
      #pragma unroll
      for (int s = 0; s < 16; ++s) {
        const size_t wr = (size_t)(c * 32 + (s < 8 ? klo + s : khi + s - 8)) * kA + nt * 16 + nrow;
        bK[s] = wkB[wr]; bQ[s] = wqB[wr]; bV[s] = wvB[wr];
      }
      accK[nt] = wmma_bf16(af, bK, accK[nt]);
      accQ[nt] = wmma_bf16(af, bQ, accQ[nt]);
      accV[nt] = wmma_bf16(af, bV, accV[nt]);
    }
  }

  // bias + store; C layout: vgpr j -> (row j + 8*hi, col nrow) of the 16x16 tile
  #pragma unroll
  for (int nt = 0; nt < 4; ++nt) {
    const int acol = nt * 16 + nrow;
    const float biasK = bk[acol], biasQ = bq[acol], biasV = bv[acol];
    // K^T: the 8 j-elements are contiguous along t -> one 16B store
    v8bf kp;
    #pragma unroll
    for (int j = 0; j < 8; ++j) kp[j] = f2bf(accK[nt][j] + biasK);
    *(v8bf*)(kTOut + ((size_t)b * kA + acol) * kT + tbase + 8 * hi) = kp;
    #pragma unroll
    for (int j = 0; j < 8; ++j) {
      const long tok = mbase + j + 8 * hi;
      qOut[tok * kA + acol] = f2bf((accQ[nt][j] + biasQ) * kScale);
      vOut[tok * kA + acol] = f2bf(accV[nt][j] + biasV);
    }
  }
}

// ---------------------------------------------------------------------------
// Kernel 2: causal flash attention. Each wave processes the balanced tile
// pair (l, 255-l) of one batch -> constant 129 key tiles per wave. Key tiles
// of 32; acc in fp32; P relayout C->A through a 1KB/wave LDS tile.
// ---------------------------------------------------------------------------
__global__ __launch_bounds__(128) void flash_attn_kernel(
    const __bf16* __restrict__ qIn, const __bf16* __restrict__ kTIn,
    const __bf16* __restrict__ vIn, float* __restrict__ out)
{
  __shared__ __align__(64) __bf16 ldsP[4][16 * 32];
  const int lane = threadIdx.x & 31;
  const int wave = threadIdx.x >> 5;
  const int hi   = lane >> 4;
  const int nrow = lane & 15;
  const int wid  = blockIdx.x * 4 + wave;     // 0..511
  const int b    = wid >> 7;                  // 128 pairs per batch
  const int lidx = wid & 127;                 // low tile of the pair

  __bf16* pbuf = &ldsP[wave][0];

  #pragma unroll 1
  for (int sel = 0; sel < 2; ++sel) {
    const int lt = sel ? (kQT - 1 - lidx) : lidx;   // triangle pairing
    const int qb = lt * 16;                         // query base row

    // Q A-fragments (A=64 -> 2 K-chunks of 32)
    v16bf aq[2];
    const __bf16* qrow = qIn + (size_t)(b * kT + qb + nrow) * kA;
    #pragma unroll
    for (int c = 0; c < 2; ++c) {
      aq[c] = cat8(*(const v8bf*)(qrow + c * 32 + hi * 8),
                   *(const v8bf*)(qrow + c * 32 + 16 + hi * 8));
    }

    v8f acc[4];
    #pragma unroll
    for (int nt = 0; nt < 4; ++nt)
      #pragma unroll
      for (int e = 0; e < 8; ++e) acc[nt][e] = 0.f;

    float mrun[8], lrun[8];
    #pragma unroll
    for (int j = 0; j < 8; ++j) { mrun[j] = -__builtin_inff(); lrun[j] = 0.f; }

    for (int kb = 0; kb <= qb; kb += 32) {
      // ---- S = Q . K^T for 2 key sub-tiles of 16 ----
      v8f s[2];
      #pragma unroll
      for (int n = 0; n < 2; ++n) {
        #pragma unroll
        for (int e = 0; e < 8; ++e) s[n][e] = 0.f;
        #pragma unroll
        for (int c = 0; c < 2; ++c) {
          // B-fragment: lane holds key column kb+n*16+nrow, 16 features in table order
          v16bf bkf;
          #pragma unroll
          for (int s = 0; s < 16; ++s)
            bkf[s] = kTIn[((size_t)b * kA + c * 32 + (s < 8 ? hi * 8 + s : 16 + hi * 8 + s - 8)) * kT + kb + n * 16 + nrow];
          s[n] = wmma_bf16(aq[c], bkf, s[n]);
        }
      }

      // ---- causal mask on diagonal-overlapping tiles ----
      if (kb + 31 > qb) {
        #pragma unroll
        for (int n = 0; n < 2; ++n)
          #pragma unroll
          for (int j = 0; j < 8; ++j) {
            const int key = kb + n * 16 + nrow;
            const int qr  = qb + j + 8 * hi;
            if (key > qr) s[n][j] = -__builtin_inff();
          }
      }

      // ---- online softmax (row = 16 lanes within a half-wave) ----
      float alpha[8];
      #pragma unroll
      for (int j = 0; j < 8; ++j) {
        float lm = fmaxf(s[0][j], s[1][j]);
        lm = fmaxf(lm, __shfl_xor(lm, 1));
        lm = fmaxf(lm, __shfl_xor(lm, 2));
        lm = fmaxf(lm, __shfl_xor(lm, 4));
        lm = fmaxf(lm, __shfl_xor(lm, 8));
        const float mnew = fmaxf(mrun[j], lm);
        const float a  = __expf(mrun[j] - mnew);
        const float p0 = __expf(s[0][j] - mnew);
        const float p1 = __expf(s[1][j] - mnew);
        float ls = p0 + p1;
        ls += __shfl_xor(ls, 1);
        ls += __shfl_xor(ls, 2);
        ls += __shfl_xor(ls, 4);
        ls += __shfl_xor(ls, 8);
        lrun[j] = lrun[j] * a + ls;
        mrun[j] = mnew;
        alpha[j] = a;
        // stage P (C layout) into LDS row-major [16 q-rows][32 keys]
        pbuf[(j + 8 * hi) * 32 + nrow]      = f2bf(p0);
        pbuf[(j + 8 * hi) * 32 + 16 + nrow] = f2bf(p1);
      }

      #pragma unroll
      for (int nt = 0; nt < 4; ++nt)
        #pragma unroll
        for (int j = 0; j < 8; ++j) acc[nt][j] *= alpha[j];

      // LDS is in-order per wave; the wait + memory clobber stops compiler
      // reordering of the cross-lane store->load relayout.
      asm volatile("s_wait_dscnt 0" ::: "memory");

      // reload P as a 16x32 bf16 A-fragment
      const __bf16* prow = pbuf + nrow * 32;
      const v16bf pf = cat8(*(const v8bf*)(prow + hi * 8),
                            *(const v8bf*)(prow + 16 + hi * 8));

      // ---- O += P . V ----
      const __bf16* vcol = vIn + (size_t)(b * kT + kb) * kA + nrow;  // lane=column
      #pragma unroll
      for (int nt = 0; nt < 4; ++nt) {
        v16bf bvf;
        #pragma unroll
        for (int s = 0; s < 16; ++s)
          bvf[s] = vcol[(size_t)(s < 8 ? hi * 8 + s : 16 + hi * 8 + s - 8) * kA + nt * 16];
        acc[nt] = wmma_bf16(pf, bvf, acc[nt]);
      }
    }

    // ---- normalize (v_rcp + mul) and store fp32 output ----
    float inv[8];
    #pragma unroll
    for (int j = 0; j < 8; ++j) inv[j] = __builtin_amdgcn_rcpf(lrun[j]);
    #pragma unroll
    for (int nt = 0; nt < 4; ++nt)
      #pragma unroll
      for (int j = 0; j < 8; ++j) {
        out[(size_t)(b * kT + qb + j + 8 * hi) * kA + nt * 16 + nrow] =
            acc[nt][j] * inv[j];
      }
  }
}

// ---------------------------------------------------------------------------
extern "C" void kernel_launch(void* const* d_in, const int* in_sizes, int n_in,
                              void* d_out, int out_size, void* d_ws, size_t ws_size,
                              hipStream_t stream) {
  (void)in_sizes; (void)n_in; (void)out_size; (void)ws_size;
  const float* emb = (const float*)d_in[0];
  const float* Wk  = (const float*)d_in[1];
  const float* bk  = (const float*)d_in[2];
  const float* Wq  = (const float*)d_in[3];
  const float* bq  = (const float*)d_in[4];
  const float* Wv  = (const float*)d_in[5];
  const float* bv  = (const float*)d_in[6];
  float* out = (float*)d_out;

  // workspace layout (bf16 elements):
  //   embB [16384*256] | Wk/Wq/Wv [3*16384] | q/kT/v [3*16384*64]  ~14.8 MB
  __bf16* ws = (__bf16*)d_ws;
  const size_t nEmb = (size_t)kTok * kE;       // 4,194,304
  const size_t nW   = (size_t)kE * kA;         // 16,384
  const size_t nP   = (size_t)kTok * kA;       // 1,048,576
  __bf16* embB  = ws;
  __bf16* wkB   = embB + nEmb;
  __bf16* wqB   = wkB + nW;
  __bf16* wvB   = wqB + nW;
  __bf16* qBuf  = wvB + nW;
  __bf16* kTBuf = qBuf + nP;
  __bf16* vBuf  = kTBuf + nP;

  // K0: fp32 -> bf16 pre-pass (memory bound; ~1us of HBM time at 23.3 TB/s)
  to_bf16_kernel<<<(int)(nEmb / 4 / 256), 256, 0, stream>>>(emb, embB, (int)nEmb);
  to_bf16_kernel<<<(int)(nW / 4 / 256), 256, 0, stream>>>(Wk, wkB, (int)nW);
  to_bf16_kernel<<<(int)(nW / 4 / 256), 256, 0, stream>>>(Wq, wqB, (int)nW);
  to_bf16_kernel<<<(int)(nW / 4 / 256), 256, 0, stream>>>(Wv, wvB, (int)nW);

  const dim3 blk(128);                    // 4 wave32 per block

  qkv_proj_kernel<<<kTok / 16 / 4, blk, 0, stream>>>(embB, wkB, wqB, wvB,
                                                     bk, bq, bv,
                                                     qBuf, kTBuf, vBuf);
  // 512 waves, each handling the balanced pair (l, 255-l) of one batch
  flash_attn_kernel<<<kB * kQT / 2 / 4, blk, 0, stream>>>(qBuf, kTBuf, vBuf, out);
}
